// SATOPModule_80015240724634
// MI455X (gfx1250) — matrix operand order repaired and checked by reading, not compiled, not run on hardware
//
#include <hip/hip_runtime.h>

// ---------------------------------------------------------------------------
// Problem constants (from reference): L=2, N=1024, B=8, F=128, H=256
// ---------------------------------------------------------------------------
#define LL 2
#define NN 1024
#define BB 8
#define FF 128
#define HH 256
#define LN_EPS 1e-5f

// WMMA vector types (probe-confirmed signatures for gfx1250)
typedef __attribute__((ext_vector_type(16))) __bf16 v16bf;
typedef __attribute__((ext_vector_type(8)))  float  v8f;

union BfFrag {            // 16 bf16 = 32 bytes = two uint4 chunks
    v16bf v;
    uint4 q[2];
};

// float -> bf16 (round to nearest even), pure integer math (no __bf16 ALU)
static __device__ __forceinline__ unsigned short f2bf(float f) {
    unsigned u = __float_as_uint(f);
    u += 0x7fffu + ((u >> 16) & 1u);
    return (unsigned short)(u >> 16);
}
static __device__ __forceinline__ unsigned pack2bf(float lo, float hi) {
    return (unsigned)f2bf(lo) | ((unsigned)f2bf(hi) << 16);
}

// ---------------------------------------------------------------------------
// CDNA5 async global->LDS DMA (ASYNCcnt-tracked).  Generic pointers to
// __shared__ truncate to the wave-relative LDS byte offset (flat aperture:
// low 32 address bits), which is exactly the VDST operand of
// GLOBAL_LOAD_ASYNC_TO_LDS_*.  GV addressing: vdst_lds, v[addr64], off.
// ---------------------------------------------------------------------------
static __device__ __forceinline__ void async_ld_b128(void* lds, const void* g) {
    unsigned l = (unsigned)(unsigned long long)lds;
    asm volatile("global_load_async_to_lds_b128 %0, %1, off"
                 :: "v"(l), "v"(g) : "memory");
}
static __device__ __forceinline__ void async_ld_b64(void* lds, const void* g) {
    unsigned l = (unsigned)(unsigned long long)lds;
    asm volatile("global_load_async_to_lds_b64 %0, %1, off"
                 :: "v"(l), "v"(g) : "memory");
}
static __device__ __forceinline__ void wait_async_le5() {
    asm volatile("s_wait_asynccnt 0x5" ::: "memory");
}
static __device__ __forceinline__ void wait_async_0() {
    asm volatile("s_wait_asynccnt 0x0" ::: "memory");
}

// ---------------------------------------------------------------------------
// Kernel S: convert Wfc (L,F,F) fp32 -> bf16 ([g][f] layout = K-contiguous rows, read as
// table-order B fragments by the fx GEMM).
// ---------------------------------------------------------------------------
__global__ void convert_wfc_kernel(const float* __restrict__ wfc,
                                   unsigned short* __restrict__ wfc_bf) {
    int i = blockIdx.x * blockDim.x + threadIdx.x;
    wfc_bf[i] = f2bf(wfc[i]);
}

// ---------------------------------------------------------------------------
// Kernel A: per-edge MLP.  a[l][n][m] = limit * tanh(W2 . elu(W1 . e + b1) + b2)
// Depends only on adjacency + weights -> computed once for both layers.
// ---------------------------------------------------------------------------
__global__ void edge_mlp_kernel(const float*  __restrict__ adj,   // [N,N,2]
                                const float*  __restrict__ W1,    // [L,H,2]
                                const float*  __restrict__ b1,    // [L,H]
                                const float*  __restrict__ W2,    // [L,1,H]
                                const float*  __restrict__ b2,    // [L,1]
                                unsigned short* __restrict__ a_bf) // [L,N,N]
{
    const int l   = blockIdx.y;
    const int tid = threadIdx.x;
    const long e  = (long)blockIdx.x * 256 + tid;

    __shared__ float2 sW1[HH];
    __shared__ float  sW2[HH];
    __shared__ float  sB1[HH];
    sW1[tid] = reinterpret_cast<const float2*>(W1)[l * HH + tid];
    sW2[tid] = W2[l * HH + tid];
    sB1[tid] = b1[l * HH + tid];
    __syncthreads();

    const float2 ed = reinterpret_cast<const float2*>(adj)[e];   // (dist, time)
    const float d = ed.x, t = ed.y;

    float acc = 0.0f;
#pragma unroll 8
    for (int k = 0; k < HH; ++k) {
        float h  = fmaf(sW1[k].x, d, fmaf(sW1[k].y, t, sB1[k]));
        float ev = h > 0.0f ? h : (__expf(h) - 1.0f);            // ELU
        acc = fmaf(sW2[k], ev, acc);
    }
    const float s  = acc + b2[l];
    const float z  = __expf(-2.0f * fabsf(s));                   // fast tanh
    float th = (1.0f - z) / (1.0f + z);
    th = copysignf(th, s);
    const float aval = (d > 0.0f) ? th : 0.0f;                   // limit mask

    a_bf[(size_t)l * NN * NN + e] = f2bf(aval);
}

// ---------------------------------------------------------------------------
// Kernel B: fx = x @ Wfc^T, stored TRANSPOSED as bf16: fxT[b][g][m].
// ---------------------------------------------------------------------------
__global__ void fx_gemm_kernel(const float*          __restrict__ x_cur,  // [B,N,F]
                               const unsigned short* __restrict__ wfc_l,  // [F,F] bf16
                               unsigned short*       __restrict__ fxT)    // [B,F,N] bf16
{
    const int b    = blockIdx.x >> 6;          // 64 m-tiles per batch
    const int mt   = blockIdx.x & 63;
    const int m0   = mt * 16;
    const int tid  = threadIdx.x;
    const int lane = tid & 31;
    const int w    = tid >> 5;                 // wave id = g-tile

    __shared__ __align__(16) unsigned short xa_tile[16 * 32];

    const int row16 = lane & 15;
    const int hiA   = (lane < 16) ? 0 : 8;     // A-frag K chunk base
    const int hiB   = (lane < 16) ? 0 : 8;     // B-frag K chunk base
    const int g     = w * 16 + row16;

    v8f c = {};
    for (int kk = 0; kk < FF; kk += 32) {
        // stage x tile -> bf16 LDS (fp32->bf16 conversion needs VALU, no DMA)
        {
            int idx = tid * 2;
            int r = idx >> 5, cidx = idx & 31;
            const float2 v = *reinterpret_cast<const float2*>(
                &x_cur[((size_t)(b * NN) + m0 + r) * FF + kk + cidx]);
            *reinterpret_cast<unsigned*>(&xa_tile[idx]) = pack2bf(v.x, v.y);
        }
        __syncthreads();

        BfFrag a, bm;
        a.q[0] = *reinterpret_cast<const uint4*>(&xa_tile[row16 * 32 + hiA]);
        a.q[1] = *reinterpret_cast<const uint4*>(&xa_tile[row16 * 32 + hiA + 16]);
        const unsigned short* bp = &wfc_l[g * FF + kk + hiB];   // two 8-K chunks (table order)
        bm.q[0] = *reinterpret_cast<const uint4*>(bp);
        bm.q[1] = *reinterpret_cast<const uint4*>(bp + 16);

        c = __builtin_amdgcn_wmma_f32_16x16x32_bf16(false, a.v, false, bm.v,
                                                    (short)0, c, false, false);
        __syncthreads();
    }

    // store transposed: lane holds column g, M = r + hiA -> 8 consecutive m
    const int mbase = m0 + hiA;
    uint4 o;
    o.x = pack2bf(c[0], c[1]);
    o.y = pack2bf(c[2], c[3]);
    o.z = pack2bf(c[4], c[5]);
    o.w = pack2bf(c[6], c[7]);
    *reinterpret_cast<uint4*>(&fxT[((size_t)(b * FF) + g) * NN + mbase]) = o;
}

// ---------------------------------------------------------------------------
// Kernel C: xh = a @ fx, + bias, + skip, + LayerNorm (fused epilogue).
// grid = (N/16, B), 256 threads (8 waves).  Block owns 16 n-rows x all 128 f.
// K loop over m in 64-chunks, DOUBLE-BUFFERED via GLOBAL_LOAD_ASYNC_TO_LDS
// (ASYNCcnt): each thread issues 5 async ops per stage (1 b64 a-tile +
// 4 b128 fxT-tile), so s_wait_asynccnt 0x5 drains exactly the previous chunk
// while the next one is in flight.  2 WMMAs per wave per chunk.
// ---------------------------------------------------------------------------
__global__ void agg_ln_kernel(const unsigned short* __restrict__ a_l,    // [N,N] bf16
                              const unsigned short* __restrict__ fxT,    // [B,F,N] bf16
                              const float*          __restrict__ x_cur,  // [B,N,F]
                              const float*          __restrict__ bias_l, // [F]
                              const float*          __restrict__ gamma_l,// [F]
                              const float*          __restrict__ beta_l, // [F]
                              float*                __restrict__ x_next) // [B,N,F]
{
    const int nt   = blockIdx.x;
    const int b    = blockIdx.y;
    const int n0   = nt * 16;
    const int tid  = threadIdx.x;
    const int lane = tid & 31;
    const int w    = tid >> 5;

    // double-buffered DMA targets + epilogue tile
    __shared__ __align__(16) unsigned short a_tile[2][16 * 64];    // 2 x 2 KB
    __shared__ __align__(16) unsigned short fx_tile[2][128 * 64];  // 2 x 16 KB
    __shared__ __align__(16) float          y_tile[16 * 128];      // 8 KB

    const int row16 = lane & 15;
    const int hiA   = (lane < 16) ? 0 : 8;
    const int hiB   = (lane < 16) ? 0 : 8;
    const int f     = w * 16 + row16;             // this lane's output column

    // per-thread staging coordinates (fixed across chunks)
    const int aidx = tid * 4;                     // 4 bf16 of the a-tile (b64)
    const int ar   = aidx >> 6, ac = aidx & 63;
    const int ffr  = tid >> 1, half = tid & 1;    // 32 bf16 of fxT row (4xb128)

    // issue one chunk's DMA into buffer `buf`
    auto stage = [&](int buf, int mt) {
        const int m0 = mt * 64;
        async_ld_b64(&a_tile[buf][aidx],
                     &a_l[(size_t)(n0 + ar) * NN + m0 + ac]);
        const unsigned short* src =
            &fxT[((size_t)(b * FF) + ffr) * NN + m0 + half * 32];
        unsigned short* dst = &fx_tile[buf][ffr * 64 + half * 32];
        async_ld_b128(dst + 0,  src + 0);
        async_ld_b128(dst + 8,  src + 8);
        async_ld_b128(dst + 16, src + 16);
        async_ld_b128(dst + 24, src + 24);
    };

    v8f c = {};
    stage(0, 0);
    for (int mt = 0; mt < NN / 64; ++mt) {
        const int buf = mt & 1;
        if (mt + 1 < NN / 64) {
            stage(buf ^ 1, mt + 1);   // prefetch next chunk into other buffer
            wait_async_le5();         // previous chunk's 5 ops have landed
        } else {
            wait_async_0();
        }
        __syncthreads();              // all waves' DMA for this chunk visible

#pragma unroll
        for (int ks = 0; ks < 64; ks += 32) {
            BfFrag a, bm;
            a.q[0]  = *reinterpret_cast<const uint4*>(
                          &a_tile[buf][row16 * 64 + ks + hiA]);
            a.q[1]  = *reinterpret_cast<const uint4*>(
                          &a_tile[buf][row16 * 64 + ks + hiA + 16]);
            bm.q[0] = *reinterpret_cast<const uint4*>(
                          &fx_tile[buf][f * 64 + ks + hiB]);
            bm.q[1] = *reinterpret_cast<const uint4*>(
                          &fx_tile[buf][f * 64 + ks + hiB + 16]);
            c = __builtin_amdgcn_wmma_f32_16x16x32_bf16(false, a.v, false, bm.v,
                                                        (short)0, c, false, false);
        }
        __syncthreads();              // done reading buf before it is re-DMA'd
    }

    // ---- epilogue: bias add, scatter tile to LDS row-major [16][128] ----
    const float bf = bias_l[f];
#pragma unroll
    for (int r = 0; r < 8; ++r)
        y_tile[(r + hiA) * FF + f] = c[r] + bf;
    __syncthreads();

    // ---- skip + LayerNorm: wave w handles rows 2w, 2w+1 (F=128 per row) ----
#pragma unroll
    for (int rr = 0; rr < 2; ++rr) {
        const int n  = n0 + w * 2 + rr;
        const int f0 = lane * 4;
        const float4 xin = *reinterpret_cast<const float4*>(
            &x_cur[((size_t)(b * NN) + n) * FF + f0]);
        float y[4];
        y[0] = xin.x + y_tile[(w * 2 + rr) * FF + f0 + 0];
        y[1] = xin.y + y_tile[(w * 2 + rr) * FF + f0 + 1];
        y[2] = xin.z + y_tile[(w * 2 + rr) * FF + f0 + 2];
        y[3] = xin.w + y_tile[(w * 2 + rr) * FF + f0 + 3];
        float s = y[0] + y[1] + y[2] + y[3];
        float q = y[0]*y[0] + y[1]*y[1] + y[2]*y[2] + y[3]*y[3];
        for (int off = 16; off > 0; off >>= 1) {        // wave32 reduction
            s += __shfl_xor(s, off, 32);
            q += __shfl_xor(q, off, 32);
        }
        const float mean = s * (1.0f / FF);
        const float var  = q * (1.0f / FF) - mean * mean;
        const float inv  = rsqrtf(var + LN_EPS);
        float4 o;
        o.x = (y[0] - mean) * inv * gamma_l[f0 + 0] + beta_l[f0 + 0];
        o.y = (y[1] - mean) * inv * gamma_l[f0 + 1] + beta_l[f0 + 1];
        o.z = (y[2] - mean) * inv * gamma_l[f0 + 2] + beta_l[f0 + 2];
        o.w = (y[3] - mean) * inv * gamma_l[f0 + 3] + beta_l[f0 + 3];
        *reinterpret_cast<float4*>(&x_next[((size_t)(b * NN) + n) * FF + f0]) = o;
    }
}

// ---------------------------------------------------------------------------
// Host-side launcher
// ---------------------------------------------------------------------------
extern "C" void kernel_launch(void* const* d_in, const int* in_sizes, int n_in,
                              void* d_out, int out_size, void* d_ws, size_t ws_size,
                              hipStream_t stream) {
    (void)in_sizes; (void)n_in; (void)out_size; (void)ws_size;

    const float* x_in  = (const float*)d_in[0];  // [B,N,F]
    const float* adj   = (const float*)d_in[1];  // [N,N,2]
    const float* W1    = (const float*)d_in[2];  // [L,H,2]
    const float* b1    = (const float*)d_in[3];  // [L,H]
    const float* W2    = (const float*)d_in[4];  // [L,1,H]
    const float* b2    = (const float*)d_in[5];  // [L,1]
    const float* Wfc   = (const float*)d_in[6];  // [L,F,F]
    const float* bias  = (const float*)d_in[7];  // [L,F]
    const float* gamma = (const float*)d_in[8];  // [L,F]
    const float* beta  = (const float*)d_in[9];  // [L,F]
    float* out = (float*)d_out;                  // [B,N,F]

    // workspace layout (all 256B-aligned offsets)
    char* ws = (char*)d_ws;
    unsigned short* a_bf   = (unsigned short*)(ws + 0);         // L*N*N bf16  = 4 MB
    unsigned short* fxT    = (unsigned short*)(ws + 4194304);   // B*F*N bf16  = 2 MB
    unsigned short* wfc_bf = (unsigned short*)(ws + 6291456);   // L*F*F bf16  = 64 KB
    float*          x_mid  = (float*)(ws + 6356992);            // B*N*F fp32  = 4 MB

    convert_wfc_kernel<<<(LL * FF * FF) / 256, 256, 0, stream>>>(Wfc, wfc_bf);
    edge_mlp_kernel<<<dim3(NN * NN / 256, LL), 256, 0, stream>>>(
        adj, W1, b1, W2, b2, a_bf);

    const float* xc = x_in;
    for (int l = 0; l < LL; ++l) {
        float* xn = (l == LL - 1) ? out : x_mid;
        fx_gemm_kernel<<<BB * (NN / 16), 256, 0, stream>>>(
            xc, wfc_bf + (size_t)l * FF * FF, fxT);
        agg_ln_kernel<<<dim3(NN / 16, BB), 256, 0, stream>>>(
            a_bf + (size_t)l * NN * NN, fxT, xc,
            bias + l * FF, gamma + l * FF, beta + l * FF, xn);
        xc = xn;
    }
}
